// NodeAlignmentHead_3891240370584
// MI455X (gfx1250) — matrix operand order repaired and checked by reading, not compiled, not run on hardware
//
// NodeAlignmentHead for MI455X (gfx1250, wave32, WMMA).
// Pipeline: prep(normalize+bf16+transpose) -> flash attn pass A (online softmax,
// att@z2, stats) -> pass B (att^T@z1 via saved stats) -> align GEMM -> MLP GEMMs
// -> BatchNorm (2-stage reduction + fused scale/shift in place on d_out).
// All matmuls via v_wmma_f32_16x16x32_bf16. Needs ~272MB of d_ws.
#include <hip/hip_runtime.h>
#include <hip/hip_bf16.h>
#include <math.h>

typedef __attribute__((ext_vector_type(16))) __bf16 v16bf;
typedef __attribute__((ext_vector_type(8)))  __bf16 bf16x8;
typedef __attribute__((ext_vector_type(8)))  float  v8f;

#define WMMA_BF16(a, b, c) \
  __builtin_amdgcn_wmma_f32_16x16x32_bf16(false, (a), false, (b), (short)0, (c), false, false)

union AFrag { v16bf v; bf16x8 h[2]; };

static __device__ __forceinline__ v16bf load_afrag(const __bf16* base, int kb) {
  // A-operand: lane holds 16 contiguous K values [2*kb, 2*kb+16)
  AFrag f;
  f.h[0] = *(const bf16x8*)(base + 2 * kb);
  f.h[1] = *(const bf16x8*)(base + 2 * kb + 8);
  return f.v;
}
static __device__ __forceinline__ v16bf load_bfrag(const __bf16* base) {
  // B-operand: lane = column, 16 contiguous K values (32B, aligned)
  return *(const v16bf*)base;
}
static __device__ __forceinline__ v8f vzero8() {
  v8f x = {0.f, 0.f, 0.f, 0.f, 0.f, 0.f, 0.f, 0.f};
  return x;
}

// ---------------------------------------------------------------------------
// Prep: per-row L2 norm stats; emit normalized bf16 rows, raw bf16 transpose,
// and raw bf16 into first half of the concat buffer. One block per (b,n) row.
__global__ __launch_bounds__(128) void k_prep(const float* __restrict__ z,
                                              __bf16* __restrict__ nb,
                                              __bf16* __restrict__ zT,
                                              __bf16* __restrict__ cat) {
  __shared__ float red[128];
  const int tid = threadIdx.x;
  const size_t row = blockIdx.x;               // b*2048 + n
  const size_t b = row >> 11, n = row & 2047;
  const float* x = z + row * 512;
  float ss = 0.f;
  for (int i = tid; i < 512; i += 128) { float v = x[i]; ss += v * v; }
  red[tid] = ss;
  __syncthreads();
  for (int s = 64; s > 0; s >>= 1) {
    if (tid < s) red[tid] += red[tid + s];
    __syncthreads();
  }
  const float inv = 1.f / fmaxf(sqrtf(red[0]), 1e-12f);
  for (int i = tid; i < 512; i += 128) {
    float v = x[i];
    nb[row * 512 + i]                      = (__bf16)(v * inv);
    zT[(b * 512 + (size_t)i) * 2048 + n]   = (__bf16)v;
    cat[row * 1024 + i]                    = (__bf16)v;
  }
}

__global__ void k_cvt(const float* __restrict__ s, __bf16* __restrict__ d, int n) {
  int i = blockIdx.x * 256 + threadIdx.x;
  if (i < n) d[i] = (__bf16)s[i];
}

// ---------------------------------------------------------------------------
// Attention pass A: for each 16-row q tile, stream 32-wide key tiles, online
// softmax, accumulate O = P @ z2 (via z2T). 8 waves, each owns a 64-wide
// d-slice; partial S reduced across waves through LDS. Saves (rowmax, rowsum).
__global__ __launch_bounds__(256) void k_attnA(const __bf16* __restrict__ n1b,
                                               const __bf16* __restrict__ n2b,
                                               const __bf16* __restrict__ z2T,
                                               __bf16* __restrict__ cat1,
                                               float* __restrict__ stats) {
  __shared__ float part[8][16][32];
  __shared__ float Sfull[16][32];
  __shared__ __align__(32) __bf16 P[16][32];
  __shared__ float scaleL[16];
  __shared__ float linv[16];

  const int tid = threadIdx.x, wave = tid >> 5, lane = tid & 31;
  const int r = lane & 15, kb = (lane >> 4) * 8, klo = (lane >> 4) * 16;
  const int col = lane & 15, rowb = (lane >> 4) * 8;
  const int b = blockIdx.y, it = blockIdx.x;
  const int dbase = wave * 64;

  const __bf16* qbase = n1b + ((size_t)b * 2048 + (size_t)it * 16) * 512;
  const v16bf qa0 = load_afrag(qbase + (size_t)r * 512 + dbase, kb);
  const v16bf qa1 = load_afrag(qbase + (size_t)r * 512 + dbase + 32, kb);

  v8f o0 = vzero8(), o1 = vzero8(), o2 = vzero8(), o3 = vzero8();
  float m_i = -3.0e38f, l_i = 0.f;   // valid in tid<16 only

  const __bf16* kvbase = n2b + (size_t)b * 2048 * 512;
  const __bf16* vtbase = z2T + (size_t)b * 512 * 2048;

  for (int mt = 0; mt < 2048; mt += 32) {
    // Prefetch next iteration's key/value tiles toward L2/WGP$.
    if (mt + 32 < 2048) {
      __builtin_prefetch(kvbase + (size_t)(mt + 32 + r) * 512 + dbase + klo, 0, 3);
      __builtin_prefetch(vtbase + (size_t)(dbase + r) * 2048 + mt + 32 + klo, 0, 3);
    }
    v8f s0 = vzero8(), s1 = vzero8();
    {
      const __bf16* kp0 = kvbase + (size_t)(mt + r) * 512 + dbase + klo;
      const __bf16* kp1 = kvbase + (size_t)(mt + 16 + r) * 512 + dbase + klo;
      // Hoist all four B fragments so the loads clause together and the
      // four WMMAs issue back-to-back behind one wait.
      v16bf b00 = load_bfrag(kp0);
      v16bf b01 = load_bfrag(kp0 + 32);
      v16bf b10 = load_bfrag(kp1);
      v16bf b11 = load_bfrag(kp1 + 32);
      s0 = WMMA_BF16(qa0, b00, s0);
      s0 = WMMA_BF16(qa1, b01, s0);
      s1 = WMMA_BF16(qa0, b10, s1);
      s1 = WMMA_BF16(qa1, b11, s1);
    }
    for (int v = 0; v < 8; ++v) {
      part[wave][rowb + v][col]      = s0[v];
      part[wave][rowb + v][16 + col] = s1[v];
    }
    __syncthreads();
    for (int e = tid; e < 512; e += 256) {
      int rr = e >> 5, cc = e & 31;
      float acc = 0.f;
      for (int w = 0; w < 8; ++w) acc += part[w][rr][cc];
      Sfull[rr][cc] = acc;
    }
    __syncthreads();
    if (tid < 16) {
      float tmax = -3.0e38f;
      for (int c = 0; c < 32; ++c) tmax = fmaxf(tmax, Sfull[tid][c]);
      float nm = fmaxf(m_i, tmax);
      float sc = __expf(m_i - nm);
      float rs = 0.f;
      for (int c = 0; c < 32; ++c) {
        float p = __expf(Sfull[tid][c] - nm);
        rs += p;
        P[tid][c] = (__bf16)p;
      }
      l_i = l_i * sc + rs;
      m_i = nm;
      scaleL[tid] = sc;
    }
    __syncthreads();
    // Issue V loads before the O rescale so VMEM overlaps the VALU work.
    const __bf16* vp = vtbase + (size_t)(dbase + r) * 2048 + mt + klo;
    v16bf vf0 = load_bfrag(vp);
    v16bf vf1 = load_bfrag(vp + 16 * 2048);
    v16bf vf2 = load_bfrag(vp + 32 * 2048);
    v16bf vf3 = load_bfrag(vp + 48 * 2048);
    const v16bf pf = load_afrag(&P[r][0], kb);
    for (int v = 0; v < 8; ++v) {
      float sc = scaleL[rowb + v];
      o0[v] *= sc; o1[v] *= sc; o2[v] *= sc; o3[v] *= sc;
    }
    o0 = WMMA_BF16(pf, vf0, o0);
    o1 = WMMA_BF16(pf, vf1, o1);
    o2 = WMMA_BF16(pf, vf2, o2);
    o3 = WMMA_BF16(pf, vf3, o3);
    __syncthreads();
  }

  if (tid < 16) {
    size_t srow = (size_t)b * 2048 + (size_t)it * 16 + tid;
    stats[srow * 2]     = m_i;
    stats[srow * 2 + 1] = l_i;
    linv[tid] = 1.f / l_i;
  }
  __syncthreads();
  const size_t orow0 = (size_t)b * 2048 + (size_t)it * 16;
  for (int v = 0; v < 8; ++v) {
    int row = rowb + v;
    float li = linv[row];
    __bf16* dst = cat1 + (orow0 + row) * 1024 + 512 + dbase;
    dst[col]      = (__bf16)(o0[v] * li);
    dst[16 + col] = (__bf16)(o1[v] * li);
    dst[32 + col] = (__bf16)(o2[v] * li);
    dst[48 + col] = (__bf16)(o3[v] * li);
  }
}

// ---------------------------------------------------------------------------
// Attention pass B: out[m,:] = sum_n att[n,m] * z1[n,:]. Recompute S^T tiles
// (roles swapped), normalize per-COLUMN with saved stats, accumulate P^T @ z1.
__global__ __launch_bounds__(256) void k_attnB(const __bf16* __restrict__ n2b,
                                               const __bf16* __restrict__ n1b,
                                               const __bf16* __restrict__ z1T,
                                               const float* __restrict__ stats,
                                               __bf16* __restrict__ cat2) {
  __shared__ float part[8][16][32];
  __shared__ __align__(32) __bf16 P[16][32];

  const int tid = threadIdx.x, wave = tid >> 5, lane = tid & 31;
  const int r = lane & 15, kb = (lane >> 4) * 8, klo = (lane >> 4) * 16;
  const int col = lane & 15, rowb = (lane >> 4) * 8;
  const int b = blockIdx.y, it = blockIdx.x;  // output m-tile
  const int dbase = wave * 64;

  const __bf16* qbase = n2b + ((size_t)b * 2048 + (size_t)it * 16) * 512;
  const v16bf qa0 = load_afrag(qbase + (size_t)r * 512 + dbase, kb);
  const v16bf qa1 = load_afrag(qbase + (size_t)r * 512 + dbase + 32, kb);

  v8f o0 = vzero8(), o1 = vzero8(), o2 = vzero8(), o3 = vzero8();
  const __bf16* kvbase = n1b + (size_t)b * 2048 * 512;
  const __bf16* vtbase = z1T + (size_t)b * 512 * 2048;
  const float* st = stats + (size_t)b * 2048 * 2;

  for (int nt = 0; nt < 2048; nt += 32) {
    if (nt + 32 < 2048) {
      __builtin_prefetch(kvbase + (size_t)(nt + 32 + r) * 512 + dbase + klo, 0, 3);
      __builtin_prefetch(vtbase + (size_t)(dbase + r) * 2048 + nt + 32 + klo, 0, 3);
    }
    v8f s0 = vzero8(), s1 = vzero8();
    {
      const __bf16* kp0 = kvbase + (size_t)(nt + r) * 512 + dbase + klo;
      const __bf16* kp1 = kvbase + (size_t)(nt + 16 + r) * 512 + dbase + klo;
      v16bf b00 = load_bfrag(kp0);
      v16bf b01 = load_bfrag(kp0 + 32);
      v16bf b10 = load_bfrag(kp1);
      v16bf b11 = load_bfrag(kp1 + 32);
      s0 = WMMA_BF16(qa0, b00, s0);
      s0 = WMMA_BF16(qa1, b01, s0);
      s1 = WMMA_BF16(qa0, b10, s1);
      s1 = WMMA_BF16(qa1, b11, s1);
    }
    for (int v = 0; v < 8; ++v) {
      part[wave][rowb + v][col]      = s0[v];
      part[wave][rowb + v][16 + col] = s1[v];
    }
    __syncthreads();
    for (int e = tid; e < 512; e += 256) {
      int rr = e >> 5, cc = e & 31;
      float acc = 0.f;
      for (int w = 0; w < 8; ++w) acc += part[w][rr][cc];
      float ms = st[(size_t)(nt + cc) * 2];
      float ls = st[(size_t)(nt + cc) * 2 + 1];
      P[rr][cc] = (__bf16)(__expf(acc - ms) / ls);
    }
    __syncthreads();
    const __bf16* vp = vtbase + (size_t)(dbase + r) * 2048 + nt + klo;
    v16bf vf0 = load_bfrag(vp);
    v16bf vf1 = load_bfrag(vp + 16 * 2048);
    v16bf vf2 = load_bfrag(vp + 32 * 2048);
    v16bf vf3 = load_bfrag(vp + 48 * 2048);
    const v16bf pf = load_afrag(&P[r][0], kb);
    o0 = WMMA_BF16(pf, vf0, o0);
    o1 = WMMA_BF16(pf, vf1, o1);
    o2 = WMMA_BF16(pf, vf2, o2);
    o3 = WMMA_BF16(pf, vf3, o3);
    __syncthreads();
  }

  const size_t orow0 = (size_t)b * 2048 + (size_t)it * 16;
  for (int v = 0; v < 8; ++v) {
    int row = rowb + v;
    __bf16* dst = cat2 + (orow0 + row) * 1024 + 512 + dbase;
    dst[col]      = (__bf16)o0[v];
    dst[16 + col] = (__bf16)o1[v];
    dst[32 + col] = (__bf16)o2[v];
    dst[48 + col] = (__bf16)o3[v];
  }
}

// ---------------------------------------------------------------------------
// GEMM: C[M,512] = A[M,K](bf16) @ W[512,K]^T + bias, optional relu; output
// either bf16 (Ob) or f32 (Of). 8 waves/block, 128x64 block tile, 32x32/wave.
__global__ __launch_bounds__(256) void k_gemm(const __bf16* __restrict__ A, int lda,
                                              const __bf16* __restrict__ W, int K,
                                              const float* __restrict__ bias,
                                              __bf16* __restrict__ Ob,
                                              float* __restrict__ Of, int relu) {
  const int tid = threadIdx.x, wave = tid >> 5, lane = tid & 31;
  const int r = lane & 15, kb = (lane >> 4) * 8, klo = (lane >> 4) * 16;
  const int wm = wave & 3, wn = wave >> 2;
  const size_t m0 = (size_t)blockIdx.x * 128 + (size_t)wm * 32;
  const int n0 = blockIdx.y * 64 + wn * 32;

  v8f acc00 = vzero8(), acc01 = vzero8(), acc10 = vzero8(), acc11 = vzero8();
  const __bf16* a0 = A + (m0 + r) * (size_t)lda;
  const __bf16* a1 = A + (m0 + 16 + r) * (size_t)lda;
  const __bf16* w0 = W + (size_t)(n0 + r) * K + klo;
  const __bf16* w1 = W + (size_t)(n0 + 16 + r) * K + klo;

  for (int k = 0; k < K; k += 32) {
    if (k + 32 < K) {
      __builtin_prefetch(a0 + k + 32, 0, 3);
      __builtin_prefetch(a1 + k + 32, 0, 3);
    }
    v16bf af0 = load_afrag(a0 + k, kb);
    v16bf af1 = load_afrag(a1 + k, kb);
    v16bf bf0 = load_bfrag(w0 + k);
    v16bf bf1 = load_bfrag(w1 + k);
    acc00 = WMMA_BF16(af0, bf0, acc00);
    acc01 = WMMA_BF16(af0, bf1, acc01);
    acc10 = WMMA_BF16(af1, bf0, acc10);
    acc11 = WMMA_BF16(af1, bf1, acc11);
  }

  const int col = lane & 15, rowb = (lane >> 4) * 8;
  const int nA = n0 + col, nB = n0 + 16 + col;
  const float bvA = bias[nA], bvB = bias[nB];
  for (int v = 0; v < 8; ++v) {
    size_t rowA = m0 + rowb + v;
    size_t rowB = m0 + 16 + rowb + v;
    float v00 = acc00[v] + bvA, v01 = acc01[v] + bvB;
    float v10 = acc10[v] + bvA, v11 = acc11[v] + bvB;
    if (relu) {
      v00 = fmaxf(v00, 0.f); v01 = fmaxf(v01, 0.f);
      v10 = fmaxf(v10, 0.f); v11 = fmaxf(v11, 0.f);
    }
    if (Ob) {
      Ob[rowA * 512 + nA] = (__bf16)v00; Ob[rowA * 512 + nB] = (__bf16)v01;
      Ob[rowB * 512 + nA] = (__bf16)v10; Ob[rowB * 512 + nB] = (__bf16)v11;
    } else {
      Of[rowA * 512 + nA] = v00; Of[rowA * 512 + nB] = v01;
      Of[rowB * 512 + nA] = v10; Of[rowB * 512 + nB] = v11;
    }
  }
}

// ---------------------------------------------------------------------------
// BatchNorm (training stats) over 32768 rows x 512 cols, two-stage reduction.
__global__ __launch_bounds__(512) void k_bnpart(const float* __restrict__ h,
                                                float* __restrict__ psum,
                                                float* __restrict__ psq) {
  const int c = threadIdx.x;
  const size_t r0 = (size_t)blockIdx.x * 256;
  float s = 0.f, q = 0.f;
  for (int i = 0; i < 256; ++i) {
    float v = h[(r0 + i) * 512 + c];
    s += v; q += v * v;
  }
  psum[blockIdx.x * 512 + c] = s;
  psq[blockIdx.x * 512 + c] = q;
}

__global__ __launch_bounds__(512) void k_bnfinal(const float* __restrict__ psum,
                                                 const float* __restrict__ psq,
                                                 const float* __restrict__ gamma,
                                                 const float* __restrict__ beta,
                                                 float* __restrict__ ab) {
  const int c = threadIdx.x;
  float s = 0.f, q = 0.f;
  for (int i = 0; i < 128; ++i) { s += psum[i * 512 + c]; q += psq[i * 512 + c]; }
  const float mu = s * (1.f / 32768.f);
  const float var = q * (1.f / 32768.f) - mu * mu;
  const float a = gamma[c] * rsqrtf(var + 1e-5f);
  ab[c * 2]     = a;
  ab[c * 2 + 1] = beta[c] - mu * a;
}

__global__ void k_bnapply(float* __restrict__ h, const float* __restrict__ ab, size_t total) {
  size_t i = (size_t)blockIdx.x * 256 + threadIdx.x;
  if (i < total) {
    int c = (int)(i & 511);
    h[i] = h[i] * ab[c * 2] + ab[c * 2 + 1];
  }
}

// ---------------------------------------------------------------------------
extern "C" void kernel_launch(void* const* d_in, const int* in_sizes, int n_in,
                              void* d_out, int out_size, void* d_ws, size_t ws_size,
                              hipStream_t stream) {
  (void)in_sizes; (void)n_in; (void)out_size; (void)ws_size;
  const float* z1      = (const float*)d_in[0];
  const float* z2      = (const float*)d_in[1];
  const float* W_align = (const float*)d_in[2];
  const float* b_align = (const float*)d_in[3];
  const float* W1      = (const float*)d_in[4];
  const float* b1      = (const float*)d_in[5];
  const float* W2      = (const float*)d_in[6];
  const float* b2      = (const float*)d_in[7];
  const float* gamma   = (const float*)d_in[8];
  const float* beta    = (const float*)d_in[9];
  float* out = (float*)d_out;

  const size_t BN = (size_t)16 * 2048;  // 32768 rows
  char* w = (char*)d_ws;
  auto carve = [&](size_t bytes) -> void* {
    void* p = (void*)w;
    w += (bytes + 255) & ~(size_t)255;
    return p;
  };
  __bf16* n1b  = (__bf16*)carve(BN * 512 * 2);      // normalized z1, bf16
  __bf16* n2b  = (__bf16*)carve(BN * 512 * 2);
  __bf16* z1T  = (__bf16*)carve(BN * 512 * 2);      // [B,512,2048] transposed raw
  __bf16* z2T  = (__bf16*)carve(BN * 512 * 2);
  __bf16* cat1 = (__bf16*)carve(BN * 1024 * 2);     // [z1 | att@z2] bf16
  __bf16* cat2 = (__bf16*)carve(BN * 1024 * 2);     // [z2 | att^T@z1] bf16
  float*  stats = (float*)carve(BN * 2 * 4);        // per-row (max, sum)
  __bf16* Wab  = (__bf16*)carve((size_t)512 * 1024 * 2);
  __bf16* W1b  = (__bf16*)carve((size_t)512 * 512 * 2);
  __bf16* W2b  = (__bf16*)carve((size_t)512 * 512 * 2);
  float*  psum0 = (float*)carve((size_t)128 * 512 * 4);
  float*  psq0  = (float*)carve((size_t)128 * 512 * 4);
  float*  psum1 = (float*)carve((size_t)128 * 512 * 4);
  float*  psq1  = (float*)carve((size_t)128 * 512 * 4);
  float*  ab0   = (float*)carve((size_t)512 * 2 * 4);
  float*  ab1   = (float*)carve((size_t)512 * 2 * 4);
  // Reuse after attention passes:
  __bf16* out1b = n1b;  __bf16* out2b = n2b;   // align outputs
  __bf16* h1b   = z1T;  __bf16* h2b   = z2T;   // MLP hidden

  // Prep + weight conversion
  k_prep<<<32768, 128, 0, stream>>>(z1, n1b, z1T, cat1);
  k_prep<<<32768, 128, 0, stream>>>(z2, n2b, z2T, cat2);
  k_cvt<<<(512 * 1024 + 255) / 256, 256, 0, stream>>>(W_align, Wab, 512 * 1024);
  k_cvt<<<(512 * 512 + 255) / 256, 256, 0, stream>>>(W1, W1b, 512 * 512);
  k_cvt<<<(512 * 512 + 255) / 256, 256, 0, stream>>>(W2, W2b, 512 * 512);

  // Flash attention, two passes
  k_attnA<<<dim3(128, 16), 256, 0, stream>>>(n1b, n2b, z2T, cat1, stats);
  k_attnB<<<dim3(128, 16), 256, 0, stream>>>(n2b, n1b, z1T, stats, cat2);

  // Align linear: [32768,1024] @ W_align^T -> bf16
  k_gemm<<<dim3(256, 8), 256, 0, stream>>>(cat1, 1024, Wab, 1024, b_align, out1b, nullptr, 0);
  k_gemm<<<dim3(256, 8), 256, 0, stream>>>(cat2, 1024, Wab, 1024, b_align, out2b, nullptr, 0);
  // MLP layer 1 (+relu) -> bf16
  k_gemm<<<dim3(256, 8), 256, 0, stream>>>(out1b, 512, W1b, 512, b1, h1b, nullptr, 1);
  k_gemm<<<dim3(256, 8), 256, 0, stream>>>(out2b, 512, W1b, 512, b1, h2b, nullptr, 1);
  // MLP layer 2 -> f32 straight into d_out (pre-BN)
  k_gemm<<<dim3(256, 8), 256, 0, stream>>>(h1b, 512, W2b, 512, b2, nullptr, out, 0);
  k_gemm<<<dim3(256, 8), 256, 0, stream>>>(h2b, 512, W2b, 512, b2, nullptr, out + BN * 512, 0);

  // BatchNorm (biased var, eps=1e-5) applied in place on d_out
  k_bnpart<<<128, 512, 0, stream>>>(out, psum0, psq0);
  k_bnpart<<<128, 512, 0, stream>>>(out + BN * 512, psum1, psq1);
  k_bnfinal<<<1, 512, 0, stream>>>(psum0, psq0, gamma, beta, ab0);
  k_bnfinal<<<1, 512, 0, stream>>>(psum1, psq1, gamma, beta, ab1);
  const size_t total = BN * 512;
  k_bnapply<<<(unsigned)((total + 255) / 256), 256, 0, stream>>>(out, ab0, total);
  k_bnapply<<<(unsigned)((total + 255) / 256), 256, 0, stream>>>(out + BN * 512, ab1, total);
}
